// ConvLayer_43173011259392
// MI455X (gfx1250) — compile-verified, matrix operand order checked by reading, not run on hardware
//
#include <hip/hip_runtime.h>
#include <hip/hip_bf16.h>

typedef __attribute__((ext_vector_type(16))) __bf16 v16bf;
typedef __attribute__((ext_vector_type(8)))  __bf16 v8bf;
typedef __attribute__((ext_vector_type(8)))  float  v8f;

union Frag { v16bf v; v8bf h[2]; };

__device__ __forceinline__ float softplus_f(float x) {
    return fmaxf(x, 0.f) + log1pf(expf(-fabsf(x)));
}

// ---------------- utility kernels ----------------

__global__ void zero_ws_k(float* __restrict__ ws, int n) {
    int i = blockIdx.x * 256 + threadIdx.x;
    if (i < n) ws[i] = 0.f;
}

// Pack W (192x128 f32, row-major [k][n]) into bf16 in the exact per-lane
// B-fragment order for v_wmma_f32_16x16x32_bf16:
// flat index o = ((ks*8 + nt)*32 + lane)*16 + j
//   n  = nt*16 + lane%16 ; hi = lane/16
//   k  = ks*32 + (j<8 ? 0 : 16) + hi*8 + (j&7)
__global__ void pack_W_k(const float* __restrict__ W, __bf16* __restrict__ Wp) {
    int o = blockIdx.x * 256 + threadIdx.x;          // 0 .. 24575
    int j    = o & 15;
    int lane = (o >> 4) & 31;
    int grp  = o >> 9;
    int nt   = grp & 7;
    int ks   = grp >> 3;
    int n    = nt * 16 + (lane & 15);
    int hi   = lane >> 4;
    int k    = ks * 32 + ((j < 8) ? 0 : 16) + hi * 8 + (j & 7);
    Wp[o] = (__bf16)W[k * 128 + n];
}

// ---------------- fused GEMM passes ----------------
// Block: 128 threads (4 waves), 64 edges per block, full 128 output cols.
// PASSB==0: GEMM -> accumulate BN1 column sums / sumsq.
// PASSB==1: GEMM -> BN1 -> sigmoid*softplus -> atomic scatter to atom_sums.

template <int PASSB>
__global__ void __launch_bounds__(128)
conv_gemm_k(const float* __restrict__ atom_in, const float* __restrict__ nbr_fea,
            const int* __restrict__ sidx, const int* __restrict__ nidx,
            const __bf16* __restrict__ Wp, const float* __restrict__ bias,
            const float* __restrict__ g1, const float* __restrict__ be1,
            float* __restrict__ s1sum, float* __restrict__ s1sq,
            const float* __restrict__ mean1, const float* __restrict__ invstd1,
            float* __restrict__ atom_sums, int E)
{
    __shared__ __align__(16) __bf16 sA[64 * 200];   // 64 rows x 192 cols, pad to 200
    __shared__ int   sS[64];
    __shared__ float sSum[128];
    __shared__ float sSq[128];

    const int t = threadIdx.x;
    const long long ebase = (long long)blockIdx.x * 64;

    // ---- stage: gather + f32->bf16 convert into LDS (2 threads / row) ----
    {
        int row = t >> 1, half = t & 1;
        long long e = ebase + row; if (e >= E) e = E - 1;
        int si = sidx[e], ni = nidx[e];
        if (half == 0) sS[row] = si;
        const float* srcs[3] = { atom_in + (long long)si * 64,
                                 atom_in + (long long)ni * 64,
                                 nbr_fea + e * 64 };
        #pragma unroll
        for (int seg = 0; seg < 3; ++seg) {
            const float4* s4 = (const float4*)(srcs[seg] + half * 32);
            __bf16* dst = &sA[row * 200 + seg * 64 + half * 32];
            #pragma unroll
            for (int q = 0; q < 8; ++q) {
                float4 v = s4[q];
                dst[4*q+0] = (__bf16)v.x; dst[4*q+1] = (__bf16)v.y;
                dst[4*q+2] = (__bf16)v.z; dst[4*q+3] = (__bf16)v.w;
            }
        }
        if (!PASSB) { sSum[t] = 0.f; sSq[t] = 0.f; }
    }
    __syncthreads();

    const int wave  = t >> 5, lane = t & 31;
    const int laneM = lane & 15, hi = lane >> 4;
    const int mbase = wave * 16;

    v8f c[8] = {};

    // ---- K loop: 6 x (1 A-frag reused over 8 N-tiles) = 48 WMMAs / wave ----
    #pragma unroll
    for (int ks = 0; ks < 6; ++ks) {
        const __bf16* ap = &sA[(mbase + laneM) * 200 + ks * 32 + hi * 8];
        Frag a;
        a.h[0] = *(const v8bf*)ap;          // K = ks*32 + hi*8      .. +7
        a.h[1] = *(const v8bf*)(ap + 16);   // K = ks*32 + 16 + hi*8 .. +7
        const __bf16* wp = Wp + ks * 8 * 512 + lane * 16;
        #pragma unroll
        for (int nt = 0; nt < 8; ++nt) {
            Frag b_;
            b_.h[0] = *(const v8bf*)(wp + nt * 512);
            b_.h[1] = *(const v8bf*)(wp + nt * 512 + 8);
            c[nt] = __builtin_amdgcn_wmma_f32_16x16x32_bf16(
                        false, a.v, false, b_.v, (short)0, c[nt], false, false);
        }
    }

    if (!PASSB) {
        // ---- BN1 statistics epilogue ----
        #pragma unroll
        for (int nt = 0; nt < 8; ++nt) {
            int col = nt * 16 + laneM;
            float bv = bias[col], s = 0.f, q = 0.f;
            #pragma unroll
            for (int v = 0; v < 8; ++v) { float x = c[nt][v] + bv; s += x; q += x * x; }
            atomicAdd(&sSum[col], s);
            atomicAdd(&sSq[col], q);
        }
        __syncthreads();
        atomicAdd(&s1sum[t], sSum[t]);
        atomicAdd(&s1sq[t],  sSq[t]);
    } else {
        // ---- BN1 + sigmoid(filter)*softplus(core) + scatter epilogue ----
        // filter col c lives in N-tile nt, core col c+64 in nt+4: same lane/vgpr.
        #pragma unroll
        for (int nt = 0; nt < 4; ++nt) {
            int cf = nt * 16 + laneM, cc = cf + 64;
            float bf_ = bias[cf], mf = mean1[cf], sf = invstd1[cf], gf = g1[cf], tf = be1[cf];
            float bc_ = bias[cc], mc = mean1[cc], sc = invstd1[cc], gc = g1[cc], tc = be1[cc];
            #pragma unroll
            for (int v = 0; v < 8; ++v) {
                int row = mbase + hi * 8 + v;
                long long e = ebase + row;
                float f = (c[nt    ][v] + bf_ - mf) * sf * gf + tf;
                float g = (c[nt + 4][v] + bc_ - mc) * sc * gc + tc;
                float msg = (1.f / (1.f + expf(-f))) * softplus_f(g);
                if (e < E)
                    atomicAdd(&atom_sums[(long long)sS[row] * 64 + cf], msg);
            }
        }
    }
}

// ---------------- BN2 statistics over atoms ----------------
__global__ void bn2_stats_k(const float* __restrict__ atom_sums,
                            float* __restrict__ s2sum, float* __restrict__ s2sq, int total)
{
    __shared__ float ss[64], sq[64];
    int t = threadIdx.x;
    if (t < 64) { ss[t] = 0.f; sq[t] = 0.f; }
    __syncthreads();
    int i = blockIdx.x * 256 + t;
    if (i < total) {
        float v = atom_sums[i];
        int col = t & 63;                 // 256 % 64 == 0 -> col == i & 63
        atomicAdd(&ss[col], v);
        atomicAdd(&sq[col], v * v);
    }
    __syncthreads();
    if (t < 64) { atomicAdd(&s2sum[t], ss[t]); atomicAdd(&s2sq[t], sq[t]); }
}

__global__ void finalize_stats_k(const float* __restrict__ sum, const float* __restrict__ sumsq,
                                 float* __restrict__ mean, float* __restrict__ invstd,
                                 float invcount, int C)
{
    int i = blockIdx.x * blockDim.x + threadIdx.x;
    if (i < C) {
        float m = sum[i] * invcount;
        mean[i] = m;
        float var = sumsq[i] * invcount - m * m;
        invstd[i] = rsqrtf(var + 1e-5f);
    }
}

// ---------------- final residual softplus ----------------
__global__ void epilogue_out_k(const float* __restrict__ atom_in, const float* __restrict__ atom_sums,
                               const float* __restrict__ mean2, const float* __restrict__ invstd2,
                               const float* __restrict__ g2, const float* __restrict__ b2,
                               float* __restrict__ out, int total)
{
    int i = blockIdx.x * 256 + threadIdx.x;
    if (i < total) {
        int col = i & 63;
        float ns = (atom_sums[i] - mean2[col]) * invstd2[col] * g2[col] + b2[col];
        out[i] = softplus_f(atom_in[i] + ns);
    }
}

extern "C" void kernel_launch(void* const* d_in, const int* in_sizes, int n_in,
                              void* d_out, int out_size, void* d_ws, size_t ws_size,
                              hipStream_t stream)
{
    const float* atom_in = (const float*)d_in[0];
    const float* nbr_fea = (const float*)d_in[1];
    const int*   sidx    = (const int*)d_in[2];
    const int*   nidx    = (const int*)d_in[3];
    const float* W       = (const float*)d_in[4];
    const float* b       = (const float*)d_in[5];
    const float* g1      = (const float*)d_in[6];
    const float* be1     = (const float*)d_in[7];
    const float* g2      = (const float*)d_in[8];
    const float* be2     = (const float*)d_in[9];

    const int N = in_sizes[0] / 64;
    const int E = in_sizes[2];

    // workspace layout (floats)
    float* ws       = (float*)d_ws;
    float* s1sum    = ws;            // 128
    float* s1sq     = ws + 128;      // 128
    float* mean1    = ws + 256;      // 128
    float* invstd1  = ws + 384;      // 128
    float* s2sum    = ws + 512;      // 64
    float* s2sq     = ws + 576;      // 64
    float* mean2    = ws + 640;      // 64
    float* invstd2  = ws + 704;      // 64
    float* atom_sums = ws + 768;     // N*64
    __bf16* Wp = (__bf16*)(ws + 768 + (size_t)N * 64);  // 192*128 bf16

    const int zn = 768 + N * 64;
    zero_ws_k<<<(zn + 255) / 256, 256, 0, stream>>>(ws, zn);
    pack_W_k<<<(192 * 128) / 256, 256, 0, stream>>>(W, Wp);

    const int gBlocks = (E + 63) / 64;
    conv_gemm_k<0><<<gBlocks, 128, 0, stream>>>(atom_in, nbr_fea, sidx, nidx, Wp, b,
                                                g1, be1, s1sum, s1sq, mean1, invstd1,
                                                atom_sums, E);
    finalize_stats_k<<<1, 128, 0, stream>>>(s1sum, s1sq, mean1, invstd1, 1.f / (float)E, 128);
    conv_gemm_k<1><<<gBlocks, 128, 0, stream>>>(atom_in, nbr_fea, sidx, nidx, Wp, b,
                                                g1, be1, s1sum, s1sq, mean1, invstd1,
                                                atom_sums, E);
    const int tot = N * 64;
    bn2_stats_k<<<(tot + 255) / 256, 256, 0, stream>>>(atom_sums, s2sum, s2sq, tot);
    finalize_stats_k<<<1, 64, 0, stream>>>(s2sum, s2sq, mean2, invstd2, 1.f / (float)N, 64);
    epilogue_out_k<<<(tot + 255) / 256, 256, 0, stream>>>(atom_in, atom_sums, mean2, invstd2,
                                                          g2, be2, (float*)d_out, tot);
}
